// VQGANTokenizer_58987080843733
// MI455X (gfx1250) — compile-verified, matrix operand order checked by reading, not run on hardware
//
#include <hip/hip_runtime.h>
#include <hip/hip_bf16.h>
#include <stdint.h>
#include <stddef.h>

// ---------------------------------------------------------------------------
// Types for CDNA5 WMMA (wave32): 16x16x32 bf16 -> f32
// ---------------------------------------------------------------------------
typedef __attribute__((ext_vector_type(16))) __bf16 v16bf;
typedef __attribute__((ext_vector_type(8)))  __bf16 v8bf;   // 16 B
typedef __attribute__((ext_vector_type(4)))  __bf16 v4bf;   // 8 B
typedef __attribute__((ext_vector_type(8)))  float  v8f;

#define DEVINL __device__ __forceinline__

DEVINL int imin(int a, int b) { return a < b ? a : b; }

// Zero a v8bf under a uniform flag using packed 32-bit selects.
DEVINL v8bf sel8(v8bf v, bool keep) {
  uint4 u = __builtin_bit_cast(uint4, v);
  u.x = keep ? u.x : 0u;
  u.y = keep ? u.y : 0u;
  u.z = keep ? u.z : 0u;
  u.w = keep ? u.w : 0u;
  return __builtin_bit_cast(v8bf, u);
}

// K-pair base position for VGPR v, lane-half h (ISA 05_wmma 16-bit A layout).
DEVINL int kmap_half(int v, int h) {
  return (v < 4) ? (2 * v + 8 * h) : (16 + 2 * (v - 4) + 8 * h);
}

// Load a 16(row) x 32(k) bf16 fragment from LDS laid out [row][k], given stride.
// Per lane this is two contiguous 16B runs -> 2x ds_load_b128.
DEVINL v16bf load_frag16x32(const __bf16* base, int stride, int lane) {
  const int m = lane & 15;
  const int h = lane >> 4;
  v16bf f;
#pragma unroll
  for (int v = 0; v < 8; ++v) {
    const int k0 = kmap_half(v, h);
    f[2 * v]     = base[m * stride + k0];
    f[2 * v + 1] = base[m * stride + k0 + 1];
  }
  return f;
}

// LDS strides (elements); 40 bf16 = 80 B, multiple of 16 B for b128 stores.
#define ASTR 40
#define BSTR 40

// ---------------------------------------------------------------------------
// conv1 only: generic implicit-GEMM conv2d. Input NCHW bf16 (Cin=3),
// natural weight layout (k = ci*KHKW + tap). Output channel-last bf16 + relu
// (Cout multiple of 32 -> packed b128 stores, no guards).
// ---------------------------------------------------------------------------
template<int KH, int KW>
__global__ __launch_bounds__(128)
void conv2d_wmma_gen(const __bf16* __restrict__ X, const __bf16* __restrict__ Wgt,
                     const float* __restrict__ Bias, __bf16* __restrict__ Y,
                     int Cin, int Hin, int Win,
                     int Cout, int Hout, int Wout,
                     int stride, int pad)
{
  __shared__ __align__(16) __bf16 As[2][32 * ASTR];
  __shared__ __align__(16) __bf16 Bs[2][64 * BSTR];
  constexpr unsigned KHKW = KH * KW;
  const int tid  = threadIdx.x;
  const int wv   = tid >> 5;
  const int lane = tid & 31;
  const int HWo  = Hout * Wout;
  const int HWi  = Hin * Win;
  const int K    = Cin * KH * KW;
  const int ksteps  = (K + 31) >> 5;
  const int pixBase = blockIdx.x * 64;
  const int coBase  = blockIdx.y * 32;

  const int am  = tid >> 2;
  const int akq = (tid & 3) * 8;
  const size_t aRowC = (size_t)imin(coBase + am, Cout - 1) * K;

  const int bcol   = tid >> 1;
  const int bkhalf = (tid & 1) * 16;
  const int bp  = pixBase + bcol;
  const int bb  = bp / HWo;
  const int brp = bp - bb * HWo;
  const int boh = brp / Wout;
  const int bow = brp - boh * Wout;
  const int ohs = boh * stride - pad;
  const int ows = bow * stride - pad;
  const __bf16* xn = X + (size_t)bb * Cin * HWi;
  const __bf16 ZB = (__bf16)0.0f;

  auto stageA = [&](int kbase, __bf16* Ad) {
    const int k = imin(kbase + akq, K - 8);
    v8bf w8 = *(const v8bf*)(Wgt + aRowC + k);
    *(v8bf*)(Ad + am * ASTR + akq) = w8;
  };

  auto stageB = [&](int kbase, __bf16* Bd) {
    const unsigned k0 = (unsigned)(kbase + bkhalf);
    const bool kvalid = (int)k0 < K;
    unsigned ci  = k0 / KHKW;
    unsigned rem = k0 - ci * KHKW;
    unsigned kh  = rem / (unsigned)KW;
    unsigned kw  = rem - kh * (unsigned)KW;
    __bf16 raw[16];
    unsigned okm = 0;
#pragma unroll
    for (int j = 0; j < 16; ++j) {
      const int ih = ohs + (int)kh;
      const int iw = ows + (int)kw;
      const bool ok = kvalid &
                      ((unsigned)ih < (unsigned)Hin) &
                      ((unsigned)iw < (unsigned)Win);
      const int off = (int)ci * HWi + ih * Win + iw;
      raw[j] = xn[ok ? off : 0];
      okm |= (unsigned)ok << j;
      ++kw;
      if (kw == KW) { kw = 0; ++kh; if (kh == KH) { kh = 0; ++ci; } }
    }
    v8bf t0, t1;
#pragma unroll
    for (int j = 0; j < 16; ++j) {
      __bf16 val = ((okm >> j) & 1u) ? raw[j] : ZB;
      if (j < 8) t0[j] = val; else t1[j - 8] = val;
    }
    *(v8bf*)(Bd + bcol * BSTR + bkhalf)     = t0;
    *(v8bf*)(Bd + bcol * BSTR + bkhalf + 8) = t1;
  };

  v8f acc[2];
#pragma unroll
  for (int g = 0; g < 2; ++g) acc[g] = (v8f){0.f,0.f,0.f,0.f,0.f,0.f,0.f,0.f};

  stageA(0, As[0]);
  stageB(0, Bs[0]);
  __syncthreads();

  for (int ks = 0; ks < ksteps; ++ks) {
    const int cur = ks & 1, nxt = cur ^ 1;
    v16bf a0 = load_frag16x32(As[cur], ASTR, lane);
    v16bf a1 = load_frag16x32(As[cur] + 16 * ASTR, ASTR, lane);
    v16bf bf = load_frag16x32(Bs[cur] + (wv * 16) * BSTR, BSTR, lane);
    if (ks + 1 < ksteps) {
      stageA((ks + 1) << 5, As[nxt]);
      stageB((ks + 1) << 5, Bs[nxt]);
    }
    acc[0] = __builtin_amdgcn_wmma_f32_16x16x32_bf16(false, a0, false, bf,
                                                     (short)0, acc[0], false, false);
    acc[1] = __builtin_amdgcn_wmma_f32_16x16x32_bf16(false, a1, false, bf,
                                                     (short)0, acc[1], false, false);
    __syncthreads();
  }

  // Channel-last packed write: co = coBase + g*16 + r + 8*h, r consecutive.
  const int n = lane & 15, h = lane >> 4;
  const int p = pixBase + wv * 16 + n;
  const size_t obase = (size_t)p * Cout + coBase + 8 * h;
#pragma unroll
  for (int g = 0; g < 2; ++g) {
    v8bf o;
#pragma unroll
    for (int r = 0; r < 8; ++r) {
      float v = acc[g][r] + Bias[coBase + g * 16 + 8 * h + r];
      v = v > 0.f ? v : 0.f;
      o[r] = (__bf16)v;
    }
    *(v8bf*)(Y + obase + g * 16) = o;
  }
}

// ---------------------------------------------------------------------------
// Tap-major implicit-GEMM conv2d, channel-last input X[b][h][w][ci].
// Weights repacked to k = tap*Cin + ci; Cin a power of two >= 16.
// Within a 16-run the tap is fixed -> the 16 ci are CONTIGUOUS in memory:
// two b128 loads + uniform zero-select per run.
// OUT_CL=true : channel-last bf16 output + relu (Cout multiple of 32).
// OUT_CL=false: NCHW f32 output, no relu, co guard (z / recon).
// ---------------------------------------------------------------------------
template<int KH, int KW, bool OUT_CL>
__global__ __launch_bounds__(128)
void conv2d_wmma_tm(const __bf16* __restrict__ X, const __bf16* __restrict__ Wgt,
                    const float* __restrict__ Bias,
                    float* __restrict__ Yf, __bf16* __restrict__ Yb,
                    int Cin, int cinShift, int Hin, int Win,
                    int Cout, int Hout, int Wout,
                    int stride, int pad)
{
  __shared__ __align__(16) __bf16 As[2][32 * ASTR];
  __shared__ __align__(16) __bf16 Bs[2][64 * BSTR];
  const int tid  = threadIdx.x;
  const int wv   = tid >> 5;
  const int lane = tid & 31;
  const int HWo  = Hout * Wout;
  const int K    = Cin * KH * KW;
  const int ksteps  = (K + 31) >> 5;
  const int pixBase = blockIdx.x * 64;
  const int coBase  = blockIdx.y * 32;

  const int am  = tid >> 2;
  const int akq = (tid & 3) * 8;
  const size_t aRowC = (size_t)imin(coBase + am, Cout - 1) * K;

  const int bcol   = tid >> 1;
  const int bkhalf = (tid & 1) * 16;
  const int bp  = pixBase + bcol;
  const int bb  = bp / HWo;
  const int brp = bp - bb * HWo;
  const int boh = brp / Wout;
  const int bow = brp - boh * Wout;
  const int ohs = boh * stride - pad;
  const int ows = bow * stride - pad;
  const __bf16* xn = X + (size_t)bb * Hin * Win * Cin;  // channel-last batch

  auto stageA = [&](int kbase, __bf16* Ad) {
    const int k = imin(kbase + akq, K - 8);
    v8bf w8 = *(const v8bf*)(Wgt + aRowC + k);
    *(v8bf*)(Ad + am * ASTR + akq) = w8;
  };

  auto stageB = [&](int kbase, __bf16* Bd) {
    const int kstart = kbase + bkhalf;           // always < K (K % 16 == 0)
    const int tap = kstart >> cinShift;          // fixed over the 16-run
    const int ci0 = kstart & (Cin - 1);
    const int kh = tap / KW;                     // constant divisor
    const int kw = tap - kh * KW;
    const int ih = ohs + kh;
    const int iw = ows + kw;
    const bool ok = ((unsigned)ih < (unsigned)Hin) &
                    ((unsigned)iw < (unsigned)Win);
    const int off = ok ? ((ih * Win + iw) * Cin + ci0) : 0;  // 16B aligned
    v8bf t0 = *(const v8bf*)(xn + off);
    v8bf t1 = *(const v8bf*)(xn + off + 8);
    t0 = sel8(t0, ok);
    t1 = sel8(t1, ok);
    *(v8bf*)(Bd + bcol * BSTR + bkhalf)     = t0;
    *(v8bf*)(Bd + bcol * BSTR + bkhalf + 8) = t1;
  };

  v8f acc[2];
#pragma unroll
  for (int g = 0; g < 2; ++g) acc[g] = (v8f){0.f,0.f,0.f,0.f,0.f,0.f,0.f,0.f};

  stageA(0, As[0]);
  stageB(0, Bs[0]);
  __syncthreads();

  for (int ks = 0; ks < ksteps; ++ks) {
    const int cur = ks & 1, nxt = cur ^ 1;
    v16bf a0 = load_frag16x32(As[cur], ASTR, lane);
    v16bf a1 = load_frag16x32(As[cur] + 16 * ASTR, ASTR, lane);
    v16bf bf = load_frag16x32(Bs[cur] + (wv * 16) * BSTR, BSTR, lane);
    if (ks + 1 < ksteps) {
      stageA((ks + 1) << 5, As[nxt]);
      stageB((ks + 1) << 5, Bs[nxt]);
    }
    acc[0] = __builtin_amdgcn_wmma_f32_16x16x32_bf16(false, a0, false, bf,
                                                     (short)0, acc[0], false, false);
    acc[1] = __builtin_amdgcn_wmma_f32_16x16x32_bf16(false, a1, false, bf,
                                                     (short)0, acc[1], false, false);
    __syncthreads();
  }

  const int n = lane & 15, h = lane >> 4;
  const int p = pixBase + wv * 16 + n;
  if (OUT_CL) {
    // channel-last packed bf16 + relu
    const size_t obase = (size_t)p * Cout + coBase + 8 * h;
#pragma unroll
    for (int g = 0; g < 2; ++g) {
      v8bf o;
#pragma unroll
      for (int r = 0; r < 8; ++r) {
        float v = acc[g][r] + Bias[coBase + g * 16 + 8 * h + r];
        v = v > 0.f ? v : 0.f;
        o[r] = (__bf16)v;
      }
      *(v8bf*)(Yb + obase + g * 16) = o;
    }
  } else {
    // NCHW f32, no relu, co guard (z and recon outputs)
    const int b  = p / HWo;
    const int rp = p - b * HWo;
    const size_t obase = (size_t)b * Cout * HWo + rp;
#pragma unroll
    for (int g = 0; g < 2; ++g) {
#pragma unroll
      for (int r = 0; r < 8; ++r) {
        const int co = coBase + g * 16 + r + 8 * h;
        if (co < Cout)
          Yf[obase + (size_t)co * HWo] = acc[g][r] + Bias[co];
      }
    }
  }
}

// ---------------------------------------------------------------------------
// ConvTranspose2d k=4 s=2 p=1 (both decoder layers: Cin=128, K=512 fixed),
// split by output parity class; weights pre-repacked per class to
// Wp[pc][co][k], k = j*128 + ci. Channel-last input and output, relu.
// ---------------------------------------------------------------------------
__global__ __launch_bounds__(128)
void tconv2d_wmma(const __bf16* __restrict__ X, const __bf16* __restrict__ Wp,
                  const float* __restrict__ Bias, __bf16* __restrict__ Yb,
                  int Hin, int Win, int Cout, int Hout, int Wout)
{
  __shared__ __align__(16) __bf16 As[2][32 * ASTR];
  __shared__ __align__(16) __bf16 Bs[2][64 * BSTR];
  constexpr int CIN = 128, K = 512, KSTEPS = 16;
  const int tid  = threadIdx.x;
  const int wv   = tid >> 5;
  const int lane = tid & 31;

  const int coBase = (blockIdx.y >> 2) * 32;
  const int pc = blockIdx.y & 3;
  const int py = pc >> 1, px = pc & 1;
  const int Ho2 = Hout >> 1, Wo2 = Wout >> 1;
  const int HW2 = Ho2 * Wo2;
  const int kh0 = (py + 1) & 1, kw0 = (px + 1) & 1;
  const int pixBase = blockIdx.x * 64;
  const __bf16* Wgt = Wp + (size_t)pc * Cout * K;   // this parity class

  const int am  = tid >> 2;
  const int akq = (tid & 3) * 8;
  const size_t aRowC = (size_t)imin(coBase + am, Cout - 1) * K;

  const int bcol   = tid >> 1;
  const int bkhalf = (tid & 1) * 16;
  const int bp  = pixBase + bcol;
  const int bb  = bp / HW2;
  const int brp = bp - bb * HW2;
  const int by  = brp / Wo2;
  const int bx  = brp - by * Wo2;
  const int t0h = 2 * by + py + 1;     // oh + 1
  const int t0w = 2 * bx + px + 1;     // ow + 1
  const __bf16* xn = X + (size_t)bb * Hin * Win * CIN;  // channel-last batch

  auto stageA = [&](int kbase, __bf16* Ad) {
    v8bf w8 = *(const v8bf*)(Wgt + aRowC + kbase + akq);   // K%32==0: no clamp
    *(v8bf*)(Ad + am * ASTR + akq) = w8;
  };

  auto stageB = [&](int kbase, __bf16* Bd) {
    const int kstart = kbase + bkhalf;       // always < K
    const int j   = kstart >> 7;             // tap, fixed over the run
    const int ci0 = kstart & (CIN - 1);
    const int kh = kh0 + 2 * (j >> 1);
    const int kw = kw0 + 2 * (j & 1);
    const int t = t0h - kh;
    const int u = t0w - kw;
    const int ih = t >> 1, iw = u >> 1;
    const bool ok = (t >= 0) & (u >= 0) & (ih < Hin) & (iw < Win);
    const int off = ok ? ((ih * Win + iw) * CIN + ci0) : 0;   // 16B aligned
    v8bf t0 = *(const v8bf*)(xn + off);
    v8bf t1 = *(const v8bf*)(xn + off + 8);
    t0 = sel8(t0, ok);
    t1 = sel8(t1, ok);
    *(v8bf*)(Bd + bcol * BSTR + bkhalf)     = t0;
    *(v8bf*)(Bd + bcol * BSTR + bkhalf + 8) = t1;
  };

  v8f acc[2];
#pragma unroll
  for (int g = 0; g < 2; ++g) acc[g] = (v8f){0.f,0.f,0.f,0.f,0.f,0.f,0.f,0.f};

  stageA(0, As[0]);
  stageB(0, Bs[0]);
  __syncthreads();

  for (int ks = 0; ks < KSTEPS; ++ks) {
    const int cur = ks & 1, nxt = cur ^ 1;
    v16bf a0 = load_frag16x32(As[cur], ASTR, lane);
    v16bf a1 = load_frag16x32(As[cur] + 16 * ASTR, ASTR, lane);
    v16bf bf = load_frag16x32(Bs[cur] + (wv * 16) * BSTR, BSTR, lane);
    if (ks + 1 < KSTEPS) {
      stageA((ks + 1) << 5, As[nxt]);
      stageB((ks + 1) << 5, Bs[nxt]);
    }
    acc[0] = __builtin_amdgcn_wmma_f32_16x16x32_bf16(false, a0, false, bf,
                                                     (short)0, acc[0], false, false);
    acc[1] = __builtin_amdgcn_wmma_f32_16x16x32_bf16(false, a1, false, bf,
                                                     (short)0, acc[1], false, false);
    __syncthreads();
  }

  // Channel-last packed write at (oh,ow) of this parity class.
  const int n = lane & 15, h = lane >> 4;
  const int p  = pixBase + wv * 16 + n;
  const int b  = p / HW2;
  const int rp = p - b * HW2;
  const int y  = rp / Wo2;
  const int x  = rp - y * Wo2;
  const int oh = 2 * y + py, ow = 2 * x + px;
  const size_t obase =
      ((size_t)(b * Hout + oh) * Wout + ow) * Cout + coBase + 8 * h;
#pragma unroll
  for (int g = 0; g < 2; ++g) {
    v8bf o;
#pragma unroll
    for (int r = 0; r < 8; ++r) {
      float v = acc[g][r] + Bias[coBase + g * 16 + 8 * h + r];
      v = v > 0.f ? v : 0.f;
      o[r] = (__bf16)v;
    }
    *(v8bf*)(Yb + obase + g * 16) = o;
  }
}

// ---------------------------------------------------------------------------
// One-shot weight repacks (f32 -> bf16 + K reordering).
// ---------------------------------------------------------------------------
// Conv weights: in w[Cout][Cin][KHKW] -> out[co][tap*Cin + ci].
__global__ void repack_conv(const float* __restrict__ W, __bf16* __restrict__ Wo,
                            int Cout, int Cin, int KHKW)
{
  const int K = Cin * KHKW;
  int idx = blockIdx.x * 256 + threadIdx.x;
  if (idx < Cout * K) {
    int co = idx / K, r = idx - co * K;
    int tap = r / Cin, ci = r - tap * Cin;
    Wo[idx] = (__bf16)W[(size_t)co * K + ci * KHKW + tap];
  }
}

// Tconv weights: in w[Cin=128][Cout][4][4] ->
// out[((pc*Cout + co)*4 + j)*128 + ci], tap (kh,kw) chosen per parity class.
__global__ void repack_tconv(const float* __restrict__ W, __bf16* __restrict__ Wo,
                             int Cout)
{
  int idx = blockIdx.x * 256 + threadIdx.x;
  if (idx < 4 * Cout * 512) {
    int ci = idx & 127;
    int t  = idx >> 7;          // (pc*Cout + co)*4 + j
    int j  = t & 3;
    int t2 = t >> 2;            // pc*Cout + co
    int co = t2 % Cout;
    int pc = t2 / Cout;
    int kh = (((pc >> 1) + 1) & 1) + 2 * (j >> 1);
    int kw = (((pc & 1) + 1) & 1) + 2 * (j & 1);
    Wo[idx] = (__bf16)W[(size_t)(ci * Cout + co) * 16 + kh * 4 + kw];
  }
}

// ---------------------------------------------------------------------------
// Codebook prep: f32 -> bf16 copy + half squared norms (0.5*||c||^2).
// ---------------------------------------------------------------------------
__global__ __launch_bounds__(128)
void cb_prep(const float* __restrict__ CB, __bf16* __restrict__ CBb,
             float* __restrict__ CN)
{
  __shared__ float red[128];
  int c = blockIdx.x, t = threadIdx.x;
  float v = CB[(size_t)c * 128 + t];
  CBb[(size_t)c * 128 + t] = (__bf16)v;
  red[t] = v * v;
  __syncthreads();
  for (int s = 64; s > 0; s >>= 1) {
    if (t < s) red[t] += red[t + s];
    __syncthreads();
  }
  if (t == 0) CN[c] = 0.5f * red[0];
}

// ---------------------------------------------------------------------------
// VQ nearest-code search: argmin||f-c||^2 == argmax(f.c - 0.5||c||^2).
// Block = 4 waves = 64 flat rows; codes streamed in 128-code LDS tiles.
// ---------------------------------------------------------------------------
__global__ __launch_bounds__(128)
void vq_argmin(const float* __restrict__ Z, const __bf16* __restrict__ CBb,
               const float* __restrict__ CN, int* __restrict__ IdxWS,
               float* __restrict__ IdxOut)
{
  __shared__ __align__(16) __bf16 Zs[64 * 136];
  __shared__ __align__(16) __bf16 Cs[128 * 136];
  __shared__ float ns[128];
  __shared__ float rbest[64 * 16];
  __shared__ int   ridx[64 * 16];
  const int tid = threadIdx.x, wv = tid >> 5, lane = tid & 31;
  const int R0 = blockIdx.x * 64;

  const float4* Z4 = (const float4*)Z;
#pragma unroll
  for (int i = 0; i < 16; ++i) {
    int e = tid + i * 128;
    int row = e >> 5, seg = e & 31;
    float4 f = Z4[(size_t)(R0 + row) * 32 + seg];
    v4bf o;
    o[0] = (__bf16)f.x; o[1] = (__bf16)f.y;
    o[2] = (__bf16)f.z; o[3] = (__bf16)f.w;
    *(v4bf*)(Zs + row * 136 + seg * 4) = o;
  }
  __syncthreads();

  v16bf a[4];
#pragma unroll
  for (int ksx = 0; ksx < 4; ++ksx)
    a[ksx] = load_frag16x32(Zs + (wv * 16) * 136 + ksx * 32, 136, lane);

  float best[8];
  int   bidx[8];
#pragma unroll
  for (int r = 0; r < 8; ++r) { best[r] = -3.4e38f; bidx[r] = 0; }

  const uint4* CB4 = (const uint4*)CBb;
  for (int c = 0; c < 16; ++c) {
    const int cbase = c << 7;
    __syncthreads();
#pragma unroll
    for (int i = 0; i < 16; ++i) {
      int e = tid + i * 128;
      int code = e >> 4, seg = e & 15;
      uint4 v = CB4[(size_t)(cbase + code) * 16 + seg];
      *(uint4*)(Cs + code * 136 + seg * 8) = v;
    }
    ns[tid] = CN[cbase + tid];
    __builtin_prefetch(CBb + (size_t)(cbase + 128) * 128 + tid * 8, 0, 0);
    __syncthreads();

#pragma unroll
    for (int sub = 0; sub < 8; ++sub) {
      v8f accv = (v8f){0.f,0.f,0.f,0.f,0.f,0.f,0.f,0.f};
#pragma unroll
      for (int ksx = 0; ksx < 4; ++ksx) {
        v16bf bf = load_frag16x32(Cs + (sub * 16) * 136 + ksx * 32, 136, lane);
        accv = __builtin_amdgcn_wmma_f32_16x16x32_bf16(false, a[ksx], false, bf,
                                                       (short)0, accv, false, false);
      }
      const int nn = lane & 15;
      const float nr = ns[sub * 16 + nn];
      const int  code = cbase + sub * 16 + nn;
#pragma unroll
      for (int r = 0; r < 8; ++r) {
        float s = accv[r] - nr;
        if (s > best[r]) { best[r] = s; bidx[r] = code; }
      }
    }
  }

  const int nn = lane & 15, h = lane >> 4;
#pragma unroll
  for (int r = 0; r < 8; ++r) {
    int row = wv * 16 + r + 8 * h;
    rbest[row * 16 + nn] = best[r];
    ridx [row * 16 + nn] = bidx[r];
  }
  __syncthreads();
  if (tid < 64) {
    float bs = -3.4e38f;
    int   bi = 0;
    for (int cc = 0; cc < 16; ++cc) {
      float s = rbest[tid * 16 + cc];
      int   ii = ridx[tid * 16 + cc];
      if (s > bs || (s == bs && ii < bi)) { bs = s; bi = ii; }
    }
    IdxWS[R0 + tid]  = bi;
    IdxOut[R0 + tid] = (float)bi;
  }
}

// ---------------------------------------------------------------------------
// Quantize gather: q_st (NCHW f32, == quantized), qbf for the decoder in
// channel-last bf16 (shift-only NCHW->NHWC transform), and
// loss = 1.25*mean((q-z)^2).
// z is [8][128][64][64]; flat row i covers 128 consecutive NCHW elements.
// ---------------------------------------------------------------------------
__global__ __launch_bounds__(128)
void vq_quant(const float* __restrict__ CB, const int* __restrict__ Idx,
              const float* __restrict__ Z, float* __restrict__ Qout,
              __bf16* __restrict__ Qb, float* __restrict__ Loss)
{
  __shared__ float red[128];
  int row = blockIdx.x, t = threadIdx.x;
  int idx = Idx[row];
  float q  = CB[(size_t)idx * 128 + t];
  float zv = Z[(size_t)row * 128 + t];
  Qout[(size_t)row * 128 + t] = q;      // q_st value == quantized (NCHW)
  // NCHW linear l -> (b, c, hw); channel-last index = (b*4096 + hw)*128 + c
  int l  = row * 128 + t;
  int b  = l >> 19;             // 128*64*64 = 524288 per batch
  int rm = l & 524287;
  int c  = rm >> 12;            // 64*64 = 4096 per channel plane
  int hw = rm & 4095;
  Qb[((size_t)(b * 4096 + hw)) * 128 + c] = (__bf16)q;
  float d = q - zv;
  red[t] = d * d;
  __syncthreads();
  for (int s = 64; s > 0; s >>= 1) {
    if (t < s) red[t] += red[t + s];
    __syncthreads();
  }
  if (t == 0) atomicAdd(Loss, red[0] * (1.25f / 4194304.0f));
}

// ---------------------------------------------------------------------------
__global__ void f32_to_bf16(const float* __restrict__ in,
                            __bf16* __restrict__ out, int n)
{
  int i = (blockIdx.x * 256 + threadIdx.x) * 4;
  if (i < n) {
    float4 f = *(const float4*)(in + i);
    v4bf o;
    o[0] = (__bf16)f.x; o[1] = (__bf16)f.y;
    o[2] = (__bf16)f.z; o[3] = (__bf16)f.w;
    *(v4bf*)(out + i) = o;
  }
}

__global__ void zero_f32(float* p, int n)
{
  int i = blockIdx.x * 256 + threadIdx.x;
  if (i < n) p[i] = 0.f;
}

// ---------------------------------------------------------------------------
extern "C" void kernel_launch(void* const* d_in, const int* in_sizes, int n_in,
                              void* d_out, int out_size, void* d_ws, size_t ws_size,
                              hipStream_t stream)
{
  (void)in_sizes; (void)n_in; (void)out_size; (void)ws_size;

  const float* x   = (const float*)d_in[0];
  const float* ew1 = (const float*)d_in[1];  const float* eb1 = (const float*)d_in[2];
  const float* ew2 = (const float*)d_in[3];  const float* eb2 = (const float*)d_in[4];
  const float* ew3 = (const float*)d_in[5];  const float* eb3 = (const float*)d_in[6];
  const float* cb  = (const float*)d_in[7];
  const float* dw1 = (const float*)d_in[8];  const float* db1 = (const float*)d_in[9];
  const float* dw2 = (const float*)d_in[10]; const float* db2 = (const float*)d_in[11];
  const float* dw3 = (const float*)d_in[12]; const float* db3 = (const float*)d_in[13];

  // Output layout: recon | q_st | indices | loss
  float* out_recon = (float*)d_out;                 // 8*3*256*256   = 1,572,864
  float* out_q     = out_recon + 1572864;           // 8*128*64*64   = 4,194,304
  float* out_idx   = out_q + 4194304;               // 8*4096        =    32,768
  float* out_loss  = out_idx + 32768;               // scalar

  // Workspace carve-up (256 B aligned blocks)
  char* ws = (char*)d_ws;
  size_t off = 0;
  auto alloc = [&](size_t bytes) -> void* {
    void* p = ws + off;
    off = (off + bytes + 255) & ~(size_t)255;
    return p;
  };
  __bf16* xbf  = (__bf16*)alloc((size_t)1572864 * 2);  // NCHW (conv1 input)
  __bf16* we1  = (__bf16*)alloc((size_t)3072 * 2);     // natural layout
  __bf16* we2  = (__bf16*)alloc((size_t)131072 * 2);   // tap-major
  __bf16* we3  = (__bf16*)alloc((size_t)147456 * 2);   // tap-major
  __bf16* wd1  = (__bf16*)alloc((size_t)262144 * 2);   // per-class tap-major
  __bf16* wd2  = (__bf16*)alloc((size_t)131072 * 2);   // per-class tap-major
  __bf16* wd3  = (__bf16*)alloc((size_t)1728 * 2);     // tap-major
  __bf16* cbbf = (__bf16*)alloc((size_t)262144 * 2);
  float*  cbn  = (float*) alloc((size_t)2048 * 4);
  __bf16* h1   = (__bf16*)alloc((size_t)8388608 * 2);  // channel-last
  __bf16* h2   = (__bf16*)alloc((size_t)4194304 * 2);  // channel-last
  float*  zbuf = (float*) alloc((size_t)4194304 * 4);  // NCHW (VQ semantics)
  __bf16* qbf  = (__bf16*)alloc((size_t)4194304 * 2);  // channel-last
  __bf16* d1b  = (__bf16*)alloc((size_t)16777216 * 2); // channel-last
  __bf16* d2b  = (__bf16*)alloc((size_t)33554432 * 2); // channel-last
  int*    idxw = (int*)   alloc((size_t)32768 * 4);

  const dim3 B128(128), B256(256);

  // --- input convert + weight repacks -------------------------------------
  f32_to_bf16<<<dim3(1572864 / 1024), B256, 0, stream>>>(x, xbf, 1572864);
  f32_to_bf16<<<dim3((3072 / 4 + 255) / 256), B256, 0, stream>>>(ew1, we1, 3072);
  repack_conv<<<dim3((131072 + 255) / 256), B256, 0, stream>>>(ew2, we2, 128, 64, 16);
  repack_conv<<<dim3((147456 + 255) / 256), B256, 0, stream>>>(ew3, we3, 128, 128, 9);
  repack_conv<<<dim3((1728   + 255) / 256), B256, 0, stream>>>(dw3, wd3, 3, 64, 9);
  repack_tconv<<<dim3((262144 + 255) / 256), B256, 0, stream>>>(dw1, wd1, 128);
  repack_tconv<<<dim3((131072 + 255) / 256), B256, 0, stream>>>(dw2, wd2, 64);
  cb_prep<<<dim3(2048), B128, 0, stream>>>(cb, cbbf, cbn);

  // --- encoder ------------------------------------------------------------
  // conv1: 3->64, k4 s2 p1, 256->128 (NCHW in, channel-last out, relu)
  conv2d_wmma_gen<4, 4><<<dim3(2048, 2), B128, 0, stream>>>(
      xbf, we1, eb1, h1, 3, 256, 256, 64, 128, 128, 2, 1);
  // conv2: 64->128, k4 s2 p1, 128->64 (CL in/out, relu)
  conv2d_wmma_tm<4, 4, true><<<dim3(512, 4), B128, 0, stream>>>(
      h1, we2, eb2, nullptr, h2, 64, 6, 128, 128, 128, 64, 64, 2, 1);
  // conv3: 128->128, k3 s1 p1 (CL in, NCHW f32 out = z, no relu)
  conv2d_wmma_tm<3, 3, false><<<dim3(512, 4), B128, 0, stream>>>(
      h2, we3, eb3, zbuf, nullptr, 128, 7, 64, 64, 128, 64, 64, 1, 1);

  // --- vector quantization ------------------------------------------------
  zero_f32<<<dim3(1), B256, 0, stream>>>(out_loss, 1);
  vq_argmin<<<dim3(512), B128, 0, stream>>>(zbuf, cbbf, cbn, idxw, out_idx);
  vq_quant<<<dim3(32768), B128, 0, stream>>>(cb, idxw, zbuf, out_q, qbf, out_loss);

  // --- decoder ------------------------------------------------------------
  // tconv1: 128->128, 64->128 (CL in/out; per-class N=32768, 4 coTiles * 4 pc)
  tconv2d_wmma<<<dim3(512, 16), B128, 0, stream>>>(
      qbf, wd1, db1, d1b, 64, 64, 128, 128, 128);
  // tconv2: 128->64, 128->256 (CL in/out; per-class N=131072, 2 coTiles * 4 pc)
  tconv2d_wmma<<<dim3(2048, 8), B128, 0, stream>>>(
      d1b, wd2, db2, d2b, 128, 128, 64, 256, 256);
  // final conv: 64->3, k3 s1 p1 (CL in, NCHW f32 out = recon, no relu)
  conv2d_wmma_tm<3, 3, false><<<dim3(8192, 1), B128, 0, stream>>>(
      d2b, wd3, db3, out_recon, nullptr, 64, 6, 256, 256, 3, 256, 256, 1, 1);
}
